// Transformer_Geo_Decoder_33818572489425
// MI455X (gfx1250) — matrix operand order repaired and checked by reading, not compiled, not run on hardware
//
#include <hip/hip_runtime.h>
#include <hip/hip_bf16.h>
#include <math.h>

// ---------------------------------------------------------------------------
// MI455X / gfx1250 implementation of the Transformer_Geo_Decoder reference.
// All dense algebra runs on the CDNA5 matrix pipe via
// __builtin_amdgcn_wmma_f32_16x16x32_f16 (f16 inputs, f32 accumulate),
// staged through double-buffered LDS tiles with vectorized fp32->fp16
// conversion (v_cvt_pk_rtz_f16_f32 + b128 LDS stores).
// Wave32 fragment layouts follow CDNA5 ISA §7.12.2.
// ---------------------------------------------------------------------------

typedef __attribute__((ext_vector_type(16))) _Float16 v16h;
typedef __attribute__((ext_vector_type(8)))  _Float16 v8h;
typedef __attribute__((ext_vector_type(4)))  _Float16 v4h;
typedef __attribute__((ext_vector_type(2)))  _Float16 v2h;
typedef __attribute__((ext_vector_type(2)))  __fp16   v2hf;  // builtin return type
typedef __attribute__((ext_vector_type(8)))  float    v8f;

#define D_MODEL 128
#define NHEADS  4
#define HD      32
#define LSEQ    512
#define NGRAPH  64
#define DFF     2048
#define TOKENS  (LSEQ * NGRAPH)   // 32768
#define FF_CHUNK 4096

#define BM 64
#define BN 64
#define BK 32
#define LDT 40        // padded LDS stride in halves (80B, 16B aligned)
#define PSTR 520      // padded stride for the softmax-P tile (1040B, 8B aligned rows)

// ---- packed f32->f16 conversion helpers (v_cvt_pk_rtz_f16_f32) -------------
__device__ inline v2h pk2(float x, float y) {
  union { v2hf a; v2h b; } u;
  u.a = __builtin_amdgcn_cvt_pkrtz(x, y);
  return u.b;
}
__device__ inline v8h pack8(const float4& a, const float4& b) {
  union { v8h v; v2h h[4]; } u;
  u.h[0] = pk2(a.x, a.y); u.h[1] = pk2(a.z, a.w);
  u.h[2] = pk2(b.x, b.y); u.h[3] = pk2(b.z, b.w);
  return u.v;
}
__device__ inline v4h pack4(const float4& a) {
  union { v4h v; v2h h[2]; } u;
  u.h[0] = pk2(a.x, a.y); u.h[1] = pk2(a.z, a.w);
  return u.v;
}

// ---- A-matrix fragment: 16x32 f16, row-major tile in LDS -------------------
// lane l: M = l&15 ; lanes 0-15 cover K {0..7,16..23}, lanes 16-31 {8..15,24..31}
__device__ inline v16h load_frag_a(const _Float16* tile, int row0, int stride, int koff) {
  const int lane = threadIdx.x & 31;
  const int r  = row0 + (lane & 15);
  const int kb = (lane & 16) ? 8 : 0;
  const _Float16* p = tile + (size_t)r * stride + koff + kb;
  v8h lo = *(const v8h*)(p);
  v8h hi = *(const v8h*)(p + 16);
  v16h f;
#pragma unroll
  for (int i = 0; i < 8; ++i) { f[i] = lo[i]; f[i + 8] = hi[i]; }
  return f;
}

// ---- B-matrix fragment: 32x16 f16, stored K-contiguous per column in LDS ---
// lane l: N = l&15 ; lanes 0-15 cover K {0..7,16..23}, lanes 16-31 {8..15,24..31}
__device__ inline v16h load_frag_b(const _Float16* tile, int col0, int stride, int koff) {
  const int lane = threadIdx.x & 31;
  const int c  = col0 + (lane & 15);
  const int kb = (lane & 16) ? 8 : 0;
  const _Float16* p = tile + (size_t)c * stride + koff + kb;
  v8h lo = *(const v8h*)(p);
  v8h hi = *(const v8h*)(p + 16);
  v16h f;
#pragma unroll
  for (int i = 0; i < 8; ++i) { f[i] = lo[i]; f[i + 8] = hi[i]; }
  return f;
}

__device__ inline float gelu_exact(float x) {
  return 0.5f * x * (1.0f + erff(x * 0.7071067811865476f));
}

__device__ inline float apply_act(float x, int act) {
  switch (act) {
    case 1:  return gelu_exact(x);
    case 2:  return fmaxf(x, 0.0f);
    case 3:  return tanhf(x);
    case 4:  return tanhf(x) * 3.14159265358979323846f;
    default: return x;
  }
}

// ---------------------------------------------------------------------------
// Tile staging: A (row-major) and W^T (K-contiguous per output column).
// Block-uniform fast path: unguarded float4 global loads + packed-f16 vector
// LDS stores. Edge path: branchless clamped-index scalar loads (cndmask, no
// exec-mask divergence).
// ---------------------------------------------------------------------------
__device__ inline void stage_tiles(const float* __restrict__ A,
                                   const float* __restrict__ W,
                                   _Float16* __restrict__ As,
                                   _Float16* __restrict__ Bs,
                                   int m0, int n0, int k0,
                                   int M, int N, int K, int tid)
{
  const bool fastA = (m0 + BM <= M) && (k0 + BK <= K) && ((K & 3) == 0);
  const bool fastB = (n0 + BN <= N) && (k0 + BK <= K) && ((N & 3) == 0);

  if (fastA) {
    // 128 threads: thread t stages row r = t>>1, K-half kh = (t&1)*16
    const int r = tid >> 1, kh = (tid & 1) * 16;
    const float4* ap = (const float4*)(A + (size_t)(m0 + r) * K + k0 + kh);
    float4 a0 = ap[0], a1 = ap[1], a2 = ap[2], a3 = ap[3];
    *(v8h*)&As[r * LDT + kh]     = pack8(a0, a1);
    *(v8h*)&As[r * LDT + kh + 8] = pack8(a2, a3);
  } else {
    for (int i = tid; i < BM * BK; i += 128) {
      int r = i >> 5, kk = i & 31;
      int gm = m0 + r, gk = k0 + kk;
      float v = A[(size_t)min(gm, M - 1) * K + min(gk, K - 1)];
      As[r * LDT + kk] = (_Float16)((gm < M && gk < K) ? v : 0.0f);
    }
  }

  if (fastB) {
    // Coalesced read of W rows; transposed b16 scatter into Bs[col][k].
#pragma unroll
    for (int it = 0; it < 4; ++it) {
      int idx = tid + it * 128;              // 0..511 float4 chunks
      int k = idx >> 4, c4 = (idx & 15) * 4;
      const float4 w4 = *(const float4*)(W + (size_t)(k0 + k) * N + n0 + c4);
      Bs[(c4 + 0) * LDT + k] = (_Float16)w4.x;
      Bs[(c4 + 1) * LDT + k] = (_Float16)w4.y;
      Bs[(c4 + 2) * LDT + k] = (_Float16)w4.z;
      Bs[(c4 + 3) * LDT + k] = (_Float16)w4.w;
    }
  } else {
    for (int i = tid; i < BN * BK; i += 128) {
      int c = i >> 5, kk = i & 31;
      int gn = n0 + c, gk = k0 + kk;
      float v = W[(size_t)min(gk, K - 1) * N + min(gn, N - 1)];
      Bs[c * LDT + kk] = (_Float16)((gn < N && gk < K) ? v : 0.0f);
    }
  }
}

// ---------------------------------------------------------------------------
// Generic WMMA GEMM: C[M,N] = act(A[M,K] @ W[K,N] + bias)
// Block = 128 threads (4 waves). Block tile 64x64, K-step 32, double-buffered
// LDS so tile t+1 stages while the matrix pipe consumes tile t.
// ---------------------------------------------------------------------------
__global__ __launch_bounds__(128)
void gemm_wmma_kernel(const float* __restrict__ A, const float* __restrict__ W,
                      const float* __restrict__ bias, float* __restrict__ C,
                      int M, int N, int K, int act)
{
  __shared__ __align__(16) _Float16 As[2][BM * LDT];
  __shared__ __align__(16) _Float16 Bs[2][BN * LDT];

  const int tid  = threadIdx.x;
  const int wave = tid >> 5;
  const int lane = tid & 31;
  const int m0 = blockIdx.y * BM;
  const int n0 = blockIdx.x * BN;

  v8f acc[4];
#pragma unroll
  for (int i = 0; i < 4; ++i)
#pragma unroll
    for (int j = 0; j < 8; ++j) acc[i][j] = 0.0f;

  const int nk = (K + BK - 1) / BK;
  stage_tiles(A, W, As[0], Bs[0], m0, n0, 0, M, N, K, tid);

  for (int t = 0; t < nk; ++t) {
    __syncthreads();   // staging of tile t complete; prior compute reads done
    if (t + 1 < nk)
      stage_tiles(A, W, As[(t + 1) & 1], Bs[(t + 1) & 1],
                  m0, n0, (t + 1) * BK, M, N, K, tid);

    const _Float16* Ac = As[t & 1];
    const _Float16* Bc = Bs[t & 1];
    // Distinct fragment registers so the ds_loads pipeline ahead of the WMMAs.
    v16h a  = load_frag_a(Ac, wave * 16, LDT, 0);
    v16h b0 = load_frag_b(Bc, 0,  LDT, 0);
    v16h b1 = load_frag_b(Bc, 16, LDT, 0);
    v16h b2 = load_frag_b(Bc, 32, LDT, 0);
    v16h b3 = load_frag_b(Bc, 48, LDT, 0);
    acc[0] = __builtin_amdgcn_wmma_f32_16x16x32_f16(false, a, false, b0, (short)0, acc[0], false, false);
    acc[1] = __builtin_amdgcn_wmma_f32_16x16x32_f16(false, a, false, b1, (short)0, acc[1], false, false);
    acc[2] = __builtin_amdgcn_wmma_f32_16x16x32_f16(false, a, false, b2, (short)0, acc[2], false, false);
    acc[3] = __builtin_amdgcn_wmma_f32_16x16x32_f16(false, a, false, b3, (short)0, acc[3], false, false);
  }

  // Epilogue: C layout -> col = lane&15, row = j + 8*(lane>=16)
  const int rbase = m0 + wave * 16 + ((lane & 16) ? 8 : 0);
#pragma unroll
  for (int nt = 0; nt < 4; ++nt) {
    int c = n0 + nt * 16 + (lane & 15);
    if (c < N) {
      float bb = bias ? bias[c] : 0.0f;
#pragma unroll
      for (int j = 0; j < 8; ++j) {
        int r = rbase + j;
        if (r < M) C[(size_t)r * N + c] = apply_act(acc[nt][j] + bb, act);
      }
    }
  }
}

// ---------------------------------------------------------------------------
// Fused WMMA attention. One wave (32 threads) per (b, h, 16-row query tile).
// qkv layout: token t = l*B + b at qkv[t*384]; Q @ +h*32, K @ +128+h*32,
// V @ +256+h*32. Output merged [token, 128].
// ---------------------------------------------------------------------------
__global__ __launch_bounds__(32)
void attn_wmma_kernel(const float* __restrict__ qkv, float* __restrict__ out,
                      int L, int B)
{
  __shared__ __align__(16) _Float16 qA[16 * LDT];
  __shared__ __align__(16) _Float16 kB[16 * LDT];
  __shared__ __align__(16) float    S[16 * LSEQ];
  __shared__ __align__(16) _Float16 pA[16 * PSTR];
  __shared__ __align__(16) _Float16 vB[32 * LDT];

  const int lane = threadIdx.x;
  const int bh = blockIdx.x;      // B*NHEADS blocks
  const int b = bh >> 2, h = bh & 3;
  const int qt = blockIdx.y;      // L/16 tiles
  const float scale = 0.17677669529663687f;  // 1/sqrt(32)

  // Stage Q tile [16 x 32] as f16: float4 loads, v4h LDS stores
#pragma unroll
  for (int it = 0; it < 4; ++it) {
    int idx = lane + it * 32;            // 0..127 float4 chunks
    int r = idx >> 3, d4 = (idx & 7) * 4;
    int l = qt * 16 + r;
    const float4 q4 = *(const float4*)(qkv + (size_t)(l * B + b) * 384 + h * 32 + d4);
    *(v4h*)&qA[r * LDT + d4] = pack4(q4);
  }
  __syncthreads();
  v16h aq = load_frag_a(qA, 0, LDT, 0);

  // S = (Q K^T) * scale, key tiles of 16
  for (int kt = 0; kt < L / 16; ++kt) {
#pragma unroll
    for (int it = 0; it < 4; ++it) {
      int idx = lane + it * 32;
      int m = idx >> 3, d4 = (idx & 7) * 4;
      int lm = kt * 16 + m;
      // B^T staging: kB[col=m][k=d] = K[m][d]  (K stored row-major)
      const float4 k4 = *(const float4*)(qkv + (size_t)(lm * B + b) * 384 + 128 + h * 32 + d4);
      *(v4h*)&kB[m * LDT + d4] = pack4(k4);
    }
    __syncthreads();
    v16h bk = load_frag_b(kB, 0, LDT, 0);
    v8f s;
#pragma unroll
    for (int j = 0; j < 8; ++j) s[j] = 0.0f;
    s = __builtin_amdgcn_wmma_f32_16x16x32_f16(false, aq, false, bk, (short)0, s,
                                               false, false);
    const int rb = (lane & 16) ? 8 : 0;
    const int c = kt * 16 + (lane & 15);
#pragma unroll
    for (int j = 0; j < 8; ++j) S[(rb + j) * LSEQ + c] = s[j] * scale;
    __syncthreads();
  }

  // Exact softmax over the key axis; lanes 0-15 own one query row each.
  if (lane < 16) {
    float4* row4 = (float4*)&S[lane * LSEQ];
    float mx = -3.4e38f;
    for (int m = 0; m < LSEQ / 4; ++m) {
      float4 v = row4[m];
      mx = fmaxf(mx, fmaxf(fmaxf(v.x, v.y), fmaxf(v.z, v.w)));
    }
    float sum = 0.0f;
    for (int m = 0; m < LSEQ / 4; ++m) {
      float4 v = row4[m];
      v.x = __expf(v.x - mx); v.y = __expf(v.y - mx);
      v.z = __expf(v.z - mx); v.w = __expf(v.w - mx);
      sum += (v.x + v.y) + (v.z + v.w);
      row4[m] = v;
    }
    const float inv = 1.0f / sum;
    v4h* prow = (v4h*)&pA[lane * PSTR];
    for (int m = 0; m < LSEQ / 4; ++m) {
      float4 v = row4[m];
      float4 p; p.x = v.x * inv; p.y = v.y * inv; p.z = v.z * inv; p.w = v.w * inv;
      prow[m] = pack4(p);
    }
  }
  __syncthreads();

  // O = P @ V, key tiles of 32
  v8f o0, o1;
#pragma unroll
  for (int j = 0; j < 8; ++j) { o0[j] = 0.0f; o1[j] = 0.0f; }
  for (int kt = 0; kt < L / 32; ++kt) {
#pragma unroll
    for (int it = 0; it < 8; ++it) {
      int idx = lane + it * 32;            // 0..255 float4 chunks
      int kk = idx >> 3, d4 = (idx & 7) * 4;
      int m = kt * 32 + kk;
      // B^T staging: vB[col=d][k=kk] = V[m][d]
      const float4 v4 = *(const float4*)(qkv + (size_t)(m * B + b) * 384 + 256 + h * 32 + d4);
      vB[(d4 + 0) * LDT + kk] = (_Float16)v4.x;
      vB[(d4 + 1) * LDT + kk] = (_Float16)v4.y;
      vB[(d4 + 2) * LDT + kk] = (_Float16)v4.z;
      vB[(d4 + 3) * LDT + kk] = (_Float16)v4.w;
    }
    __syncthreads();
    v16h ap = load_frag_a(pA, 0, PSTR, kt * 32);
    v16h b0 = load_frag_b(vB, 0, LDT, 0);
    v16h b1 = load_frag_b(vB, 16, LDT, 0);
    o0 = __builtin_amdgcn_wmma_f32_16x16x32_f16(false, ap, false, b0, (short)0, o0,
                                                false, false);
    o1 = __builtin_amdgcn_wmma_f32_16x16x32_f16(false, ap, false, b1, (short)0, o1,
                                                false, false);
    __syncthreads();
  }

  const int rb = (lane & 16) ? 8 : 0;
  const int c = lane & 15;
#pragma unroll
  for (int j = 0; j < 8; ++j) {
    int l = qt * 16 + rb + j;
    size_t base = (size_t)(l * B + b) * D_MODEL + h * 32;
    out[base + c]      = o0[j];
    out[base + 16 + c] = o1[j];
  }
}

// ---------------------------------------------------------------------------
// Residual + LayerNorm (optionally fused GELU on the output). Wave per row.
// ---------------------------------------------------------------------------
__global__ __launch_bounds__(128)
void add_ln_kernel(const float* __restrict__ a, const float* __restrict__ r,
                   const float* __restrict__ g, const float* __restrict__ beta,
                   float* __restrict__ out, int M, int gelu_out)
{
  const int wave = threadIdx.x >> 5, lane = threadIdx.x & 31;
  const int row = blockIdx.x * 4 + wave;
  if (row >= M) return;
  const float4 a4 = *(const float4*)(a + (size_t)row * D_MODEL + lane * 4);
  const float4 r4 = *(const float4*)(r + (size_t)row * D_MODEL + lane * 4);
  float v[4] = {a4.x + r4.x, a4.y + r4.y, a4.z + r4.z, a4.w + r4.w};
  float s = (v[0] + v[1]) + (v[2] + v[3]);
  for (int off = 16; off; off >>= 1) s += __shfl_xor(s, off, 32);
  const float mean = s * (1.0f / 128.0f);
  float vs = 0.0f;
#pragma unroll
  for (int i = 0; i < 4; ++i) { float d = v[i] - mean; vs += d * d; }
  for (int off = 16; off; off >>= 1) vs += __shfl_xor(vs, off, 32);
  const float rstd = rsqrtf(vs * (1.0f / 128.0f) + 1e-5f);
  const float4 g4 = *(const float4*)(g + lane * 4);
  const float4 b4 = *(const float4*)(beta + lane * 4);
  float4 o;
  o.x = (v[0] - mean) * rstd * g4.x + b4.x;
  o.y = (v[1] - mean) * rstd * g4.y + b4.y;
  o.z = (v[2] - mean) * rstd * g4.z + b4.z;
  o.w = (v[3] - mean) * rstd * g4.w + b4.w;
  if (gelu_out) {
    o.x = gelu_exact(o.x); o.y = gelu_exact(o.y);
    o.z = gelu_exact(o.z); o.w = gelu_exact(o.w);
  }
  *(float4*)(out + (size_t)row * D_MODEL + lane * 4) = o;
}

// batch is sorted (repeat of arange): first-occurrence index = segment offset
__global__ void offsets_kernel(const int* __restrict__ batch, int* __restrict__ offs,
                               int N)
{
  int i = blockIdx.x * 256 + threadIdx.x;
  if (i < N) {
    if (i == 0 || batch[i] != batch[i - 1]) offs[batch[i]] = i;
  }
}

__global__ __launch_bounds__(32)
void scatter_kernel(const float* __restrict__ h, const int* __restrict__ batch,
                    const int* __restrict__ offs, float* __restrict__ X,
                    int N, int B)
{
  int t = blockIdx.x;
  if (t >= N) return;
  int g = batch[t];
  int pos = t - offs[g];
  const float4* src = (const float4*)(h + (size_t)t * D_MODEL);
  float4* dst = (float4*)(X + (size_t)(pos * B + g) * D_MODEL);
  dst[threadIdx.x] = src[threadIdx.x];
}

// DynamicTanh: y = g*tanh(alpha*x) + b  (alpha is a device scalar)
__global__ __launch_bounds__(256)
void dyt_kernel(const float4* __restrict__ x4, const float* __restrict__ alpha,
                const float* __restrict__ g, const float* __restrict__ b,
                float4* __restrict__ y4, int n4)
{
  const float a = *alpha;
  for (int i = blockIdx.x * 256 + threadIdx.x; i < n4; i += gridDim.x * 256) {
    const int c = (i * 4) & (D_MODEL - 1);
    float4 v = x4[i];
    float4 o;
    o.x = g[c + 0] * tanhf(a * v.x) + b[c + 0];
    o.y = g[c + 1] * tanhf(a * v.y) + b[c + 1];
    o.z = g[c + 2] * tanhf(a * v.z) + b[c + 2];
    o.w = g[c + 3] * tanhf(a * v.w) + b[c + 3];
    y4[i] = o;
  }
}

// L2-normalize z[L,B,32] over the B axis (dim=1), in place.
__global__ __launch_bounds__(256)
void norm_dim1_kernel(float* __restrict__ z, int L, int B)
{
  __shared__ float red[8][32];
  const int l = blockIdx.x;
  const int c = threadIdx.x & 31;
  const int grp = threadIdx.x >> 5;  // 0..7
  float s = 0.0f;
  for (int b = grp; b < B; b += 8) {
    float v = z[(size_t)(l * B + b) * 32 + c];
    s += v * v;
  }
  red[grp][c] = s;
  __syncthreads();
  if (grp == 0) {
    float t = 0.0f;
#pragma unroll
    for (int i = 0; i < 8; ++i) t += red[i][c];
    red[0][c] = 1.0f / (sqrtf(t) + 1e-10f);
  }
  __syncthreads();
  const float inv = red[0][c];
  for (int b = grp; b < B; b += 8) {
    z[(size_t)(l * B + b) * 32 + c] *= inv;
  }
}

// ---------------------------------------------------------------------------
static inline void launch_gemm(const float* A, const float* W, const float* bias,
                               float* C, int M, int N, int K, int act,
                               hipStream_t s)
{
  dim3 g((N + BN - 1) / BN, (M + BM - 1) / BM);
  gemm_wmma_kernel<<<g, 128, 0, s>>>(A, W, bias, C, M, N, K, act);
}

// Input flattening assumed jax-pytree style: dict keys sorted, lists in order.
// Top level: batch, max_len, num_graphs, params(flattened), x
enum {
  I_BATCH = 0, I_MAXLEN, I_NGRAPH,
  I_AN1B, I_AN1W, I_AN2B, I_AN2W, I_AN3B, I_AN3W,
  I_DYTA, I_DYTB, I_DYTG,
  I_IN1B, I_IN1W, I_IN2B, I_IN2W,
  I_L0 = 16,           // 12 entries per layer
  I_LIN1B = 40, I_LIN1W, I_LIN2B, I_LIN2W, I_LIN3B, I_LIN3W,
  I_RT1B, I_RT1W, I_RT2B, I_RT2W, I_RT3B, I_RT3W,
  I_X = 52
};
enum { LP_FF1B = 0, LP_FF1W, LP_FF2B, LP_FF2W, LP_LN1B, LP_LN1G,
       LP_LN2B, LP_LN2G, LP_OUTB, LP_OUTW, LP_QKVB, LP_QKVW };

extern "C" void kernel_launch(void* const* d_in, const int* in_sizes, int n_in,
                              void* d_out, int out_size, void* d_ws, size_t ws_size,
                              hipStream_t stream)
{
  (void)n_in; (void)out_size; (void)ws_size;
  const int*   batch = (const int*)d_in[I_BATCH];
  const float* x     = (const float*)d_in[I_X];
  const int Ntok = in_sizes[I_BATCH];       // 24704
  const int L = LSEQ, B = NGRAPH, T = TOKENS;

  auto F = [&](int i) { return (const float*)d_in[i]; };

  // ---- workspace carve (everything we read is written each call) ----
  char* ws = (char*)d_ws;
  auto carve = [&](size_t bytes) {
    char* p = ws;
    ws += (bytes + 255) & ~(size_t)255;
    return p;
  };
  float* h0a  = (float*)carve((size_t)Ntok * D_MODEL * 4);
  float* h0b  = (float*)carve((size_t)Ntok * D_MODEL * 4);
  int*   offs = (int*)  carve(NGRAPH * sizeof(int));
  float* X    = (float*)carve((size_t)T * D_MODEL * 4);
  float* QKV  = (float*)carve((size_t)T * 384 * 4);
  float* ATTN = (float*)carve((size_t)T * D_MODEL * 4);
  float* T1   = (float*)carve((size_t)T * D_MODEL * 4);
  float* H1   = (float*)carve((size_t)T * D_MODEL * 4);
  float* FF   = (float*)carve((size_t)FF_CHUNK * DFF * 4);

  float* RT  = (float*)d_out;                 // [512,64,7]
  float* ANG = RT + (size_t)T * 7;            // [512,64,3]

  // ---- input projection: tanh(gelu(x@in1 + b)@in2 + b) ----
  launch_gemm(x,   F(I_IN1W), F(I_IN1B), h0a, Ntok, D_MODEL, 10,      1, stream);
  launch_gemm(h0a, F(I_IN2W), F(I_IN2B), h0b, Ntok, D_MODEL, D_MODEL, 3, stream);

  // ---- scatter ragged -> dense zero-padded [L, B, D] ----
  offsets_kernel<<<(Ntok + 255) / 256, 256, 0, stream>>>(batch, offs, Ntok);
  (void)hipMemsetAsync(X, 0, (size_t)T * D_MODEL * 4, stream);
  scatter_kernel<<<Ntok, 32, 0, stream>>>(h0b, batch, offs, X, Ntok, B);

  // ---- transformer encoder layers ----
  for (int layer = 0; layer < 2; ++layer) {
    const int lb = I_L0 + layer * 12;
    launch_gemm(X, F(lb + LP_QKVW), F(lb + LP_QKVB), QKV, T, 3 * D_MODEL, D_MODEL,
                0, stream);
    {
      dim3 g(B * NHEADS, L / 16);
      attn_wmma_kernel<<<g, 32, 0, stream>>>(QKV, ATTN, L, B);
    }
    launch_gemm(ATTN, F(lb + LP_OUTW), F(lb + LP_OUTB), T1, T, D_MODEL, D_MODEL,
                0, stream);
    add_ln_kernel<<<T / 4, 128, 0, stream>>>(X, T1, F(lb + LP_LN1G),
                                             F(lb + LP_LN1B), H1, T, 0);
    // FF in M-chunks to bound the [*, 2048] intermediate
    for (int c0 = 0; c0 < T; c0 += FF_CHUNK) {
      launch_gemm(H1 + (size_t)c0 * D_MODEL, F(lb + LP_FF1W), F(lb + LP_FF1B),
                  FF, FF_CHUNK, DFF, D_MODEL, 2, stream);
      launch_gemm(FF, F(lb + LP_FF2W), F(lb + LP_FF2B),
                  T1 + (size_t)c0 * D_MODEL, FF_CHUNK, D_MODEL, DFF, 0, stream);
    }
    add_ln_kernel<<<T / 4, 128, 0, stream>>>(H1, T1, F(lb + LP_LN2G),
                                             F(lb + LP_LN2B), X, T, 1);
  }

  // ---- decoder head: DyT -> 128 -> 64 -> 32, normalize over dim=1 ----
  float* Z0 = ATTN;  // free
  dyt_kernel<<<2048, 256, 0, stream>>>((const float4*)X, F(I_DYTA), F(I_DYTG),
                                       F(I_DYTB), (float4*)Z0,
                                       T * D_MODEL / 4);
  float* Z1 = QKV;   // free, large enough
  launch_gemm(Z0, F(I_LIN1W), F(I_LIN1B), Z1, T, 128, 128, 1, stream);
  float* Z2 = T1;    // free
  launch_gemm(Z1, F(I_LIN2W), F(I_LIN2B), Z2, T, 64, 128, 1, stream);
  float* Z3 = H1;    // free
  launch_gemm(Z2, F(I_LIN3W), F(I_LIN3B), Z3, T, 32, 64, 0, stream);
  norm_dim1_kernel<<<L, 256, 0, stream>>>(Z3, L, B);

  // ---- rt head -> d_out[0 : T*7] ----
  float* R1 = ATTN;
  float* R2 = QKV;
  launch_gemm(Z3, F(I_RT1W), F(I_RT1B), R1, T, 64, 32, 1, stream);
  launch_gemm(R1, F(I_RT2W), F(I_RT2B), R2, T, 32, 64, 1, stream);
  launch_gemm(R2, F(I_RT3W), F(I_RT3B), RT, T, 7, 32, 0, stream);

  // ---- angle head -> d_out[T*7 : T*10], tanh * pi ----
  launch_gemm(Z3, F(I_AN1W), F(I_AN1B), R1, T, 64, 32, 1, stream);
  launch_gemm(R1, F(I_AN2W), F(I_AN2B), R2, T, 32, 64, 1, stream);
  launch_gemm(R2, F(I_AN3W), F(I_AN3B), ANG, T, 3, 32, 4, stream);
}
